// AttnTeacherTargetMerger_18691697672676
// MI455X (gfx1250) — matrix operand order repaired and checked by reading, not compiled, not run on hardware
//
#include <hip/hip_runtime.h>

typedef __attribute__((ext_vector_type(16))) _Float16 v16h;
typedef __attribute__((ext_vector_type(8)))  _Float16 v8h;
typedef __attribute__((ext_vector_type(8)))  float    v8f;

namespace {
constexpr int kC   = 5;        // classes / seq len
constexpr int kE   = 64;       // embed dim
constexpr int kH   = 160;
constexpr int kW   = 160;
constexpr int kHW  = kH * kW;  // 25600
constexpr int kPix = 32;       // pixels per workgroup
constexpr int kRows = kPix * kC;   // 160 rows (row = p*5 + cls)
constexpr int kStr  = 72;          // padded LDS row stride in halfs (16B aligned rows)
constexpr int kThreads = 256;      // 8 wave32s
constexpr int kBlocks  = 102400 / kPix;  // 3200
constexpr int kTileB   = kRows * kStr * 2;   // 23040 bytes per [160][72] f16 tile

// ---- LDS layout (byte offsets). Buffers aliased by lifetime; every reuse is
// separated by an existing __syncthreads(). Total ~156 KB (< 320 KB/WGP). ----
constexpr size_t oW_B   = 0;                       // Wq|Wk|Wv|Wo rows: 4*64*64 f16 = 32768
constexpr size_t oCls_B = oW_B + 32768;            // cls_w padded 16x320 f16 = 10240
constexpr size_t oAt_B  = oCls_B + 10240;          // conv(teacher) [160][72] f16 (live to stage 4)
constexpr size_t oAg_B  = oAt_B + kTileB;          // conv(target); REUSED as ctx after QKV
constexpr size_t oQ_B   = oAg_B + kTileB;          // Q;  Q+K region REUSED as f32 res [160][64]
constexpr size_t oK_B   = oQ_B + kTileB;           // K
constexpr size_t oV_B   = oK_B + kTileB;           // V;  REUSED as LN output flat [32][320] f16
constexpr size_t oXt_B  = oV_B + kTileB;           // teacher pixels [32][5] f32
constexpr size_t oXg_B  = oXt_B + kPix * kC * 4;   // target  pixels [32][5] f32
constexpr size_t kSmemBytes = oXg_B + kPix * kC * 4;   // 159488 bytes
} // namespace

// A-fragment: 16x32 f16 (M x K). lanes 0-15 -> K base 0, lanes 16-31 -> K base 8;
// v16h elems 0..7 = K kb..kb+7, elems 8..15 = K kb+16..kb+23.
__device__ __forceinline__ v16h ldA(const _Float16* tile, int stride, int lane, int kOff) {
  const int m  = lane & 15;
  const int kb = (lane < 16) ? 0 : 8;
  const _Float16* p = tile + m * stride + kOff + kb;
  v8h lo = *(const v8h*)(p);
  v8h hi = *(const v8h*)(p + 16);
  v16h r;
#pragma unroll
  for (int i = 0; i < 8; ++i) { r[i] = lo[i]; r[i + 8] = hi[i]; }
  return r;
}

// B-fragment: 32x16 f16 (K x N), B = W^T so column n of B = row n of W (contiguous).
// lanes 0-15 -> K base 0, lanes 16-31 -> K base 8; elems 0..7 = K kb..kb+7, elems 8..15 = K kb+16..kb+23.
__device__ __forceinline__ v16h ldB(const _Float16* wrows, int stride, int lane, int kOff) {
  const int n  = lane & 15;
  const int kb = (lane < 16) ? 0 : 8;
  const _Float16* p = wrows + n * stride + kOff + kb;
  v8h lo = *(const v8h*)(p);
  v8h hi = *(const v8h*)(p + 16);
  v16h r;
#pragma unroll
  for (int i = 0; i < 8; ++i) { r[i] = lo[i]; r[i + 8] = hi[i]; }
  return r;
}

__device__ __forceinline__ v8f wmma_f16(v16h a, v16h b, v8f c) {
  return __builtin_amdgcn_wmma_f32_16x16x32_f16(false, a, false, b, (short)0, c, false, false);
}

// One [160x64] @ [64x64]^T projection: 40 output tiles, 5 per wave (uniform),
// 2 k-steps of WMMA 16x16x32, bias fused into the f16 writeback.
__device__ __forceinline__ void proj_gemm(const _Float16* __restrict__ A,
                                          const _Float16* __restrict__ Wm,
                                          const float* __restrict__ bias,
                                          _Float16* __restrict__ D,
                                          int wave, int lane) {
  for (int t = wave; t < 40; t += 8) {
    int mt = t >> 2, nt = t & 3;
    v8f acc = {};
#pragma unroll
    for (int ks = 0; ks < 2; ++ks) {
      v16h a = ldA(A + mt * 16 * kStr, kStr, lane, ks * 32);
      v16h b = ldB(Wm + nt * 16 * 64, 64, lane, ks * 32);
      acc = wmma_f16(a, b, acc);
    }
    int col = nt * 16 + (lane & 15);
    float bi = bias[col];
    int rbase = mt * 16 + ((lane < 16) ? 0 : 8);
#pragma unroll
    for (int r = 0; r < 8; ++r)
      D[(rbase + r) * kStr + col] = (_Float16)(acc[r] + bi);
  }
}

__global__ __launch_bounds__(kThreads)
void attn_merger_fused(const float* __restrict__ teacher,
                       const float* __restrict__ target,
                       const float* __restrict__ teacher_w,
                       const float* __restrict__ target_w,
                       const float* __restrict__ in_proj_w,   // [192][64]
                       const float* __restrict__ in_proj_b,   // [192]
                       const float* __restrict__ out_w,       // [64][64]
                       const float* __restrict__ out_b,       // [64]
                       const float* __restrict__ ln_g,
                       const float* __restrict__ ln_b,
                       const float* __restrict__ cls_w,       // [5][320]
                       const float* __restrict__ cls_b,       // [5]
                       float* __restrict__ seg)               // [4][5][160][160]
{
  extern __shared__ __align__(16) char smem[];
  _Float16* sW   = (_Float16*)(smem + oW_B);     // Wq @0, Wk @4096, Wv @8192, Wo @12288 (halfs)
  _Float16* sCls = (_Float16*)(smem + oCls_B);
  _Float16* sAt  = (_Float16*)(smem + oAt_B);
  _Float16* sAg  = (_Float16*)(smem + oAg_B);
  _Float16* sQ   = (_Float16*)(smem + oQ_B);
  _Float16* sK   = (_Float16*)(smem + oK_B);
  _Float16* sV   = (_Float16*)(smem + oV_B);
  _Float16* sCtx = sAg;                          // alias: conv(target) dead after QKV
  float*    sRes = (float*)(smem + oQ_B);        // alias: Q/K dead after attention
  _Float16* sRF  = sV;                           // alias: V dead after attention
  float* sXt = (float*)(smem + oXt_B);
  float* sXg = (float*)(smem + oXg_B);

  const int tid  = threadIdx.x;
  const int lane = tid & 31;
  const int wave = tid >> 5;
  const int blk  = blockIdx.x;

  // ---- Stage 0: stage weights (f32 -> f16) and the 32x5 input pixels into LDS ----
  for (int i = tid; i < 16384; i += kThreads)
    sW[i] = (_Float16)(i < 12288 ? in_proj_w[i] : out_w[i - 12288]);
  for (int i = tid; i < 16 * 320; i += kThreads) {
    int row = i / 320;
    sCls[i] = (_Float16)(row < kC ? cls_w[i] : 0.0f);
  }
  for (int i = tid; i < 2 * kPix * kC; i += kThreads) {
    int which = i / (kPix * kC);
    int j = i - which * (kPix * kC);
    int p = j / kC, g = j - p * kC;
    int n = blk * kPix + p;
    int b = n / kHW, rem = n - b * kHW;
    float v = (which == 0 ? teacher : target)[(b * kC + g) * kHW + rem];
    (which == 0 ? sXt : sXg)[j] = v;
  }
  __syncthreads();

  // ---- Stage 1: grouped 1x1 conv + E-major rearrange:
  //      conv[p,cls,e] = w[e*5+cls] * x[p, (e*5+cls)/64] ----
  for (int idx = tid; idx < kRows * kE; idx += kThreads) {
    int r = idx >> 6, e = idx & 63;
    int p = r / kC, cls = r - p * kC;
    int ch = e * kC + cls;
    int g  = ch >> 6;
    sAt[r * kStr + e] = (_Float16)(teacher_w[ch] * sXt[p * kC + g]);
    sAg[r * kStr + e] = (_Float16)(target_w[ch]  * sXg[p * kC + g]);
  }
  __syncthreads();

  // ---- Stage 2: Q/K/V projections as three uniform WMMA GEMMs (pointer-invariant loops) ----
  proj_gemm(sAt, sW,        in_proj_b,       sQ, wave, lane);
  proj_gemm(sAg, sW + 4096, in_proj_b + 64,  sK, wave, lane);
  proj_gemm(sAg, sW + 8192, in_proj_b + 128, sV, wave, lane);
  __syncthreads();

  // ---- Stage 3: attention core (32 heads, head_dim 2, seq 5) in VALU, f32 softmax ----
  {
    const int p  = tid >> 3;       // pixel 0..31
    const int hg = tid & 7;        // 4 heads each
    const float scale = 0.70710678118654752f;  // 1/sqrt(2)
#pragma unroll
    for (int hh = 0; hh < 4; ++hh) {
      int h = hg * 4 + hh, c0 = 2 * h;
      float q[5][2], k[5][2], v[5][2];
#pragma unroll
      for (int s = 0; s < kC; ++s) {
        int row = (p * kC + s) * kStr;
        q[s][0] = (float)sQ[row + c0]; q[s][1] = (float)sQ[row + c0 + 1];
        k[s][0] = (float)sK[row + c0]; k[s][1] = (float)sK[row + c0 + 1];
        v[s][0] = (float)sV[row + c0]; v[s][1] = (float)sV[row + c0 + 1];
      }
      float ctx[5][2];
#pragma unroll
      for (int sq = 0; sq < kC; ++sq) {
        float sc[5], mx = -3.4e38f;
#pragma unroll
        for (int sk = 0; sk < kC; ++sk) {
          sc[sk] = (q[sq][0] * k[sk][0] + q[sq][1] * k[sk][1]) * scale;
          mx = fmaxf(mx, sc[sk]);
        }
        float sum = 0.f;
#pragma unroll
        for (int sk = 0; sk < kC; ++sk) { sc[sk] = __expf(sc[sk] - mx); sum += sc[sk]; }
        float inv = 1.0f / sum, o0 = 0.f, o1 = 0.f;
#pragma unroll
        for (int sk = 0; sk < kC; ++sk) {
          float a = sc[sk] * inv;
          o0 += a * v[sk][0]; o1 += a * v[sk][1];
        }
        ctx[sq][0] = o0; ctx[sq][1] = o1;
      }
      // write context after all Q/K/V of this head were consumed (sCtx aliases sAg, not QKV)
#pragma unroll
      for (int sq = 0; sq < kC; ++sq) {
        int row = (p * kC + sq) * kStr;
        sCtx[row + c0]     = (_Float16)ctx[sq][0];
        sCtx[row + c0 + 1] = (_Float16)ctx[sq][1];
      }
    }
  }
  __syncthreads();

  // ---- Stage 4: out-proj [160x64]@[64x64]^T (WMMA) + bias + residual -> f32 sRes ----
  for (int t = wave; t < 40; t += 8) {            // 5 tiles per wave (uniform)
    int mt = t >> 2, nt = t & 3;
    v8f acc = {};
#pragma unroll
    for (int ks = 0; ks < 2; ++ks) {
      v16h a = ldA(sCtx + mt * 16 * kStr, kStr, lane, ks * 32);
      v16h b = ldB(sW + 12288 + nt * 16 * 64, 64, lane, ks * 32);
      acc = wmma_f16(a, b, acc);
    }
    int col = nt * 16 + (lane & 15);
    float bias = out_b[col];
    int rbase = mt * 16 + ((lane < 16) ? 0 : 8);
#pragma unroll
    for (int r = 0; r < 8; ++r) {
      int row = rbase + r;
      sRes[row * kE + col] = acc[r] + bias + (float)sAt[row * kStr + col];
    }
  }
  __syncthreads();

  // ---- Stage 5: LayerNorm over E=64 per row; write per-pixel flat [32][320] f16 ----
  if (tid < kRows) {
    const float* rp = sRes + tid * kE;
    float s = 0.f, s2 = 0.f;
#pragma unroll
    for (int e = 0; e < kE; ++e) { float x = rp[e]; s += x; s2 += x * x; }
    float mu = s * (1.0f / kE);
    float var = s2 * (1.0f / kE) - mu * mu;
    float rstd = rsqrtf(var + 1e-5f);
    int p = tid / kC, cls = tid - p * kC;
    _Float16* dst = sRF + p * 320 + cls * kE;
#pragma unroll
    for (int e = 0; e < kE; ++e)
      dst[e] = (_Float16)((rp[e] - mu) * rstd * ln_g[e] + ln_b[e]);
  }
  __syncthreads();

  // ---- Stage 6: classifier head [32x320]@[5x320]^T via WMMA (N padded to 16), scatter to seg ----
  if (wave < 2) {                                  // wave-uniform: EXEC all ones inside
    int mt = wave;
    v8f acc = {};
#pragma unroll
    for (int ks = 0; ks < 10; ++ks) {
      v16h a = ldA(sRF + mt * 16 * 320, 320, lane, ks * 32);
      v16h b = ldB(sCls,                320, lane, ks * 32);
      acc = wmma_f16(a, b, acc);
    }
    int ncls = lane & 15;
    if (ncls < kC) {
      float cb = cls_b[ncls];
      int rbase = mt * 16 + ((lane < 16) ? 0 : 8);
#pragma unroll
      for (int r = 0; r < 8; ++r) {
        int n = blk * kPix + rbase + r;
        int b = n / kHW, rem = n - b * kHW;
        seg[(b * kC + ncls) * kHW + rem] = acc[r] + cb;
      }
    }
  }
}

extern "C" void kernel_launch(void* const* d_in, const int* in_sizes, int n_in,
                              void* d_out, int out_size, void* d_ws, size_t ws_size,
                              hipStream_t stream) {
  (void)in_sizes; (void)n_in; (void)out_size; (void)d_ws; (void)ws_size;
  const float* teacher   = (const float*)d_in[0];
  const float* target    = (const float*)d_in[1];
  const float* teacher_w = (const float*)d_in[2];
  const float* target_w  = (const float*)d_in[3];
  const float* in_proj_w = (const float*)d_in[4];
  const float* in_proj_b = (const float*)d_in[5];
  const float* out_w     = (const float*)d_in[6];
  const float* out_b     = (const float*)d_in[7];
  const float* ln_g      = (const float*)d_in[8];
  const float* ln_b      = (const float*)d_in[9];
  const float* cls_w     = (const float*)d_in[10];
  const float* cls_b     = (const float*)d_in[11];
  float* seg = (float*)d_out;

  attn_merger_fused<<<kBlocks, kThreads, kSmemBytes, stream>>>(
      teacher, target, teacher_w, target_w, in_proj_w, in_proj_b,
      out_w, out_b, ln_g, ln_b, cls_w, cls_b, seg);
}
